// LabelSentenceAttention_2173253451826
// MI455X (gfx1250) — matrix operand order repaired and checked by reading, not compiled, not run on hardware
//
#include <hip/hip_runtime.h>
#include <hip/hip_bf16.h>

#define SEQ   32768
#define H2    1024
#define DA    512
#define NC    16

typedef __attribute__((ext_vector_type(16))) __bf16 v16bf;
typedef __attribute__((ext_vector_type(8)))  float  v8f;
typedef __attribute__((ext_vector_type(2)))  float  f32x2;
typedef __attribute__((ext_vector_type(2)))  __bf16 bf16x2;

__device__ __forceinline__ bf16x2 cvt2bf(float x, float y) {
  f32x2 v; v.x = x; v.y = y;
  return __builtin_convertvector(v, bf16x2);   // -> v_cvt_pk_bf16_f32
}

__device__ __forceinline__ v8f wmma_bf16(v16bf a, v16bf b, v8f c) {
  return __builtin_amdgcn_wmma_f32_16x16x32_bf16(false, a, false, b, (short)0, c, false, false);
}

// A-matrix (16x32 bf16) K-offset for VGPR-pair p (elements 2p,2p+1), per ISA layout
__device__ __forceinline__ int ka_off(int p, int half) {
  return 2 * p + (p >= 4 ? 8 : 0) + 8 * half;
}
// B-matrix (32x16 bf16) K-offset for VGPR-pair p
__device__ __forceinline__ int kb_off(int p, int half) {
  return 2 * p + (p >= 4 ? 8 : 0) + 8 * half;
}

__device__ __forceinline__ float fast_tanh(float x) {
#if __has_builtin(__builtin_amdgcn_tanhf)
  return __builtin_amdgcn_tanhf(x);   // native V_TANH_F32 on gfx1250
#else
  return tanhf(x);
#endif
}

__device__ __forceinline__ float sigmoidf_(float x) { return 1.0f / (1.0f + __expf(-x)); }

// ---------------------------------------------------------------------------
// Kernel P: one-time weight conversion.
//   W1t[n*H2+k]  = bf16(W1[k*DA+n])   (512 x 1024, transposed: B k-pairs contiguous)
//   W2t[c*DA+k]  = bf16(W2[k*NC+c])   (16 x 512, transposed)
//   labb[c*DA+k] = bf16(lab[c*DA+k])  (16 x 512, as-is: A k-pairs contiguous)
// ---------------------------------------------------------------------------
__global__ void k_prep(const float* __restrict__ W1, const float* __restrict__ W2,
                       const float* __restrict__ lab,
                       __bf16* __restrict__ W1t, __bf16* __restrict__ W2t,
                       __bf16* __restrict__ labb) {
  int i = blockIdx.x * blockDim.x + threadIdx.x;
  if (i < DA * H2) {
    int n = i >> 10, k = i & (H2 - 1);
    W1t[i] = (__bf16)W1[(size_t)k * DA + n];
  } else if (i < DA * H2 + NC * DA) {
    int j = i - DA * H2;
    int c = j >> 9, k = j & (DA - 1);
    W2t[j] = (__bf16)W2[k * NC + c];
  } else if (i < DA * H2 + 2 * NC * DA) {
    int j = i - DA * H2 - NC * DA;
    labb[j] = (__bf16)lab[j];
  }
}

// ---------------------------------------------------------------------------
// Kernel 0: zero accumulators
// ---------------------------------------------------------------------------
__global__ void k_zero(float* p, int n) {
  int i = blockIdx.x * blockDim.x + threadIdx.x;
  if (i < n) p[i] = 0.0f;
}

// ---------------------------------------------------------------------------
// Kernel 1: scores = tanh(sentence @ W1 + b1) @ W2 + b2        (SEQ x 16)
// One block per 32 sentence rows (2 M-tiles: B fragments reused across both,
// halving W1t traffic). 8 waves; LDS-staged bf16 A tile, H aliased onto it.
// ---------------------------------------------------------------------------
__global__ __launch_bounds__(256) void k_gemm1(const float* __restrict__ sent,
                                               const __bf16* __restrict__ W1t,
                                               const float* __restrict__ b1,
                                               const __bf16* __restrict__ W2t,
                                               const float* __restrict__ b2,
                                               float* __restrict__ scores) {
  __shared__ __bf16 As[32 * H2];        // 64 KB (32 rows are one contiguous global block)
  __bf16* Hs = As;                      // reused after the K loop (32 x 512 = 32 KB)
  const int row0 = blockIdx.x * 32;
  const int tid  = threadIdx.x;

  {
    const float* base = sent + (size_t)row0 * H2;
    for (int i = tid * 4; i < 32 * H2; i += 256 * 4) {
      float4 v = *(const float4*)(base + i);
      *(bf16x2*)(As + i)     = cvt2bf(v.x, v.y);
      *(bf16x2*)(As + i + 2) = cvt2bf(v.z, v.w);
    }
  }
  __syncthreads();

  const int wave = tid >> 5;
  const int lane = tid & 31;
  const int half = lane >> 4;
  const int lrow = lane & 15;

  // H columns [wave*64, wave*64+64) for both 16-row M-tiles
  v8f acc0[4] = {}, acc1[4] = {};
  for (int kk = 0; kk < H2; kk += 32) {
    v16bf a0, a1;
#pragma unroll
    for (int p = 0; p < 8; p++) {
      int ko = kk + ka_off(p, half);
      bf16x2 t0 = *(const bf16x2*)(As + lrow * H2 + ko);
      bf16x2 t1 = *(const bf16x2*)(As + (16 + lrow) * H2 + ko);
      a0[2 * p] = t0.x; a0[2 * p + 1] = t0.y;
      a1[2 * p] = t1.x; a1[2 * p + 1] = t1.y;
    }
#pragma unroll
    for (int t = 0; t < 4; t++) {
      int nb = wave * 64 + t * 16;
      v16bf b;
#pragma unroll
      for (int p = 0; p < 8; p++) {
        bf16x2 w = *(const bf16x2*)(W1t + (size_t)(nb + lrow) * H2 + kk + kb_off(p, half));
        b[2 * p] = w.x; b[2 * p + 1] = w.y;
      }
      acc0[t] = wmma_bf16(a0, b, acc0[t]);
      acc1[t] = wmma_bf16(a1, b, acc1[t]);
    }
  }
  __syncthreads();   // all waves done reading As before H overwrites it

  // bias + tanh -> bf16 H tile in LDS (row-major 32 x 512)
#pragma unroll
  for (int t = 0; t < 4; t++) {
    int n = wave * 64 + t * 16 + lrow;
    float bias = b1[n];
#pragma unroll
    for (int v = 0; v < 8; v++) {
      int m = v + 8 * half;
      Hs[m * DA + n]        = (__bf16)fast_tanh(acc0[t][v] + bias);
      Hs[(16 + m) * DA + n] = (__bf16)fast_tanh(acc1[t][v] + bias);
    }
  }
  __syncthreads();

  if (wave < 2) {  // waves 0,1 each handle one 16-row M-tile; EXEC all ones per wave
    const __bf16* Hrow = Hs + (size_t)(wave * 16) * DA;
    v8f acc2 = {};
    for (int kk = 0; kk < DA; kk += 32) {
      v16bf a, b;
#pragma unroll
      for (int p = 0; p < 8; p++) {
        bf16x2 t = *(const bf16x2*)(Hrow + lrow * DA + kk + ka_off(p, half));
        a[2 * p] = t.x; a[2 * p + 1] = t.y;
        bf16x2 w = *(const bf16x2*)(W2t + lrow * DA + kk + kb_off(p, half));
        b[2 * p] = w.x; b[2 * p + 1] = w.y;
      }
      acc2 = wmma_bf16(a, b, acc2);
    }
    float bias2 = b2[lrow];
#pragma unroll
    for (int v = 0; v < 8; v++) {
      int m = v + 8 * half;
      scores[(size_t)(row0 + wave * 16 + m) * NC + lrow] = acc2[v] + bias2;
    }
  }
}

// ---------------------------------------------------------------------------
// Kernel 2: per-class softmax stats over the sequence axis
// stats[0..15] = max_c, stats[16..31] = 1 / sum(exp(x - max))
// ---------------------------------------------------------------------------
__global__ __launch_bounds__(256) void k_colstats(const float* __restrict__ scores,
                                                  float* __restrict__ stats) {
  const int c = blockIdx.x;
  __shared__ float red[256];
  float m = -3.4e38f;
  for (int s = threadIdx.x; s < SEQ; s += 256) m = fmaxf(m, scores[(size_t)s * NC + c]);
  red[threadIdx.x] = m;
  __syncthreads();
  for (int o = 128; o > 0; o >>= 1) {
    if (threadIdx.x < o) red[threadIdx.x] = fmaxf(red[threadIdx.x], red[threadIdx.x + o]);
    __syncthreads();
  }
  float cmax = red[0];
  __syncthreads();
  float sum = 0.0f;
  for (int s = threadIdx.x; s < SEQ; s += 256) sum += __expf(scores[(size_t)s * NC + c] - cmax);
  red[threadIdx.x] = sum;
  __syncthreads();
  for (int o = 128; o > 0; o >>= 1) {
    if (threadIdx.x < o) red[threadIdx.x] += red[threadIdx.x + o];
    __syncthreads();
  }
  if (threadIdx.x == 0) {
    stats[c]      = cmax;
    stats[16 + c] = 1.0f / red[0];
  }
}

// ---------------------------------------------------------------------------
// Kernel 3: self_att (16 x 1024) += softmax(scores)^T @ sentence  (per S-chunk)
// Block handles 256 sequence rows; softmax weights computed ONCE per block into
// a bank-padded LDS buffer; wave w covers columns [w*128, w*128+128).
// ---------------------------------------------------------------------------
#define WSTRIDE 258   // 256 + 2 pad: breaks 16-way LDS bank conflicts, keeps pairs aligned

__global__ __launch_bounds__(256) void k_selfatt(const float* __restrict__ sent,
                                                 const float* __restrict__ scores,
                                                 const float* __restrict__ stats,
                                                 float* __restrict__ self_att) {
  __shared__ __bf16 wbuf[NC * WSTRIDE];   // ~8.1 KB, wbuf[c][s_local]
  __shared__ float cmaxs[NC], cinvs[NC];
  const int s0  = blockIdx.x * 256;
  const int tid = threadIdx.x;
  if (tid < NC) { cmaxs[tid] = stats[tid]; cinvs[tid] = stats[16 + tid]; }
  __syncthreads();

  {  // thread t computes the 16 normalized weights of sequence row s0+t
    const float4* sp = (const float4*)(scores + (size_t)(s0 + tid) * NC);
    float4 q0 = sp[0], q1 = sp[1], q2 = sp[2], q3 = sp[3];
    float v[16] = {q0.x, q0.y, q0.z, q0.w, q1.x, q1.y, q1.z, q1.w,
                   q2.x, q2.y, q2.z, q2.w, q3.x, q3.y, q3.z, q3.w};
#pragma unroll
    for (int c = 0; c < NC; c++)
      wbuf[c * WSTRIDE + tid] = (__bf16)(__expf(v[c] - cmaxs[c]) * cinvs[c]);
  }
  __syncthreads();

  const int wave = tid >> 5;
  const int lane = tid & 31;
  const int half = lane >> 4;
  const int lrow = lane & 15;   // = class row of the A matrix for this lane

  v8f acc[8] = {};
  for (int kk = 0; kk < 256; kk += 32) {
    v16bf a;
#pragma unroll
    for (int p = 0; p < 8; p++) {
      bf16x2 t = *(const bf16x2*)(wbuf + lrow * WSTRIDE + kk + ka_off(p, half));
      a[2 * p] = t.x; a[2 * p + 1] = t.y;
    }
#pragma unroll
    for (int t = 0; t < 8; t++) {
      int nb = wave * 128 + t * 16;
      v16bf b;
#pragma unroll
      for (int p = 0; p < 8; p++) {
        int k = s0 + kk + kb_off(p, half);
        float x0 = sent[(size_t)k * H2 + nb + lrow];
        float x1 = sent[(size_t)(k + 1) * H2 + nb + lrow];
        bf16x2 w = cvt2bf(x0, x1);
        b[2 * p] = w.x; b[2 * p + 1] = w.y;
      }
      acc[t] = wmma_bf16(a, b, acc[t]);
    }
  }
#pragma unroll
  for (int t = 0; t < 8; t++) {
    int n = wave * 128 + t * 16 + lrow;
#pragma unroll
    for (int v = 0; v < 8; v++) {
      int m = v + 8 * half;
      atomicAdd(&self_att[m * H2 + n], acc[t][v]);
    }
  }
}

// ---------------------------------------------------------------------------
// Kernel 4: label attention, fused per 128-row S-chunk:
//   m1 = label_embed @ h1^T, m2 = label_embed @ h2^T   (16 x 128 each, in LDS)
//   label_att[:, 0:512]  += m1 @ h1 ; label_att[:, 512:1024] += m2 @ h2
// ---------------------------------------------------------------------------
#define MSTRIDE 130   // 128 + 2 pad

__global__ __launch_bounds__(256) void k_labelatt(const float* __restrict__ sent,
                                                  const __bf16* __restrict__ labb,
                                                  float* __restrict__ label_att) {
  __shared__ __bf16 m1buf[NC * MSTRIDE];
  __shared__ __bf16 m2buf[NC * MSTRIDE];
  const int s0   = blockIdx.x * 128;
  const int wave = threadIdx.x >> 5;
  const int lane = threadIdx.x & 31;
  const int half = lane >> 4;
  const int lrow = lane & 15;

  // Stage 1: wave w computes m1/m2 for s-columns [16w, 16w+16), K = 512
  const size_t srow = (size_t)(s0 + wave * 16 + lrow) * H2;  // B-col lane's sentence row
  v8f am1 = {}, am2 = {};
  for (int kk = 0; kk < DA; kk += 32) {
    v16bf a, b1v, b2v;
#pragma unroll
    for (int p = 0; p < 8; p++) {
      bf16x2 ta = *(const bf16x2*)(labb + lrow * DA + kk + ka_off(p, half));
      a[2 * p] = ta.x; a[2 * p + 1] = ta.y;
      int kb = kk + kb_off(p, half);
      f32x2 x1 = *(const f32x2*)(sent + srow + kb);        // contiguous pair
      f32x2 x2 = *(const f32x2*)(sent + srow + DA + kb);   // contiguous pair
      bf16x2 w1p = __builtin_convertvector(x1, bf16x2);
      bf16x2 w2p = __builtin_convertvector(x2, bf16x2);
      b1v[2 * p] = w1p.x; b1v[2 * p + 1] = w1p.y;
      b2v[2 * p] = w2p.x; b2v[2 * p + 1] = w2p.y;
    }
    am1 = wmma_bf16(a, b1v, am1);
    am2 = wmma_bf16(a, b2v, am2);
  }
#pragma unroll
  for (int v = 0; v < 8; v++) {
    int m = v + 8 * half;
    int n = wave * 16 + lrow;
    m1buf[m * MSTRIDE + n] = (__bf16)am1[v];
    m2buf[m * MSTRIDE + n] = (__bf16)am2[v];
  }
  __syncthreads();

  // Stage 2: wave w covers output columns [64w, 64w+64), K = 128 (the chunk)
#pragma unroll
  for (int t = 0; t < 4; t++) {
    int nb = wave * 64 + t * 16;
    const __bf16* mbuf = (nb < DA) ? m1buf : m2buf;
    v8f acc = {};
    for (int kk = 0; kk < 128; kk += 32) {
      v16bf a, b;
#pragma unroll
      for (int p = 0; p < 8; p++) {
        bf16x2 ta = *(const bf16x2*)(mbuf + lrow * MSTRIDE + kk + ka_off(p, half));
        a[2 * p] = ta.x; a[2 * p + 1] = ta.y;
        int kb = kk + kb_off(p, half);  // local s index; sentence col nb+lrow spans h1|h2
        float x0 = sent[(size_t)(s0 + kb) * H2 + nb + lrow];
        float x1 = sent[(size_t)(s0 + kb + 1) * H2 + nb + lrow];
        bf16x2 w = cvt2bf(x0, x1);
        b[2 * p] = w.x; b[2 * p + 1] = w.y;
      }
      acc = wmma_bf16(a, b, acc);
    }
#pragma unroll
    for (int v = 0; v < 8; v++) {
      int m = v + 8 * half;
      atomicAdd(&label_att[m * H2 + nb + lrow], acc[v]);
    }
  }
}

// ---------------------------------------------------------------------------
// Kernel 5: gates, doc average, output head
// ---------------------------------------------------------------------------
__device__ __forceinline__ float block_reduce_sum(float v, float* red) {
  int tid = threadIdx.x;
  red[tid] = v;
  __syncthreads();
  for (int o = 128; o > 0; o >>= 1) {
    if (tid < o) red[tid] += red[tid + o];
    __syncthreads();
  }
  float r = red[0];
  __syncthreads();
  return r;
}

__global__ __launch_bounds__(256) void k_final(const float* __restrict__ self_att,
                                               const float* __restrict__ label_att,
                                               const float* __restrict__ w1w,
                                               const float* __restrict__ w1b,
                                               const float* __restrict__ w2w,
                                               const float* __restrict__ w2b,
                                               const float* __restrict__ Wout,
                                               const float* __restrict__ bout,
                                               float* __restrict__ out) {
  __shared__ float red[256];
  __shared__ float g1n[NC];
  __shared__ float avg[H2];
  const int tid = threadIdx.x;

  for (int c = 0; c < NC; c++) {
    float p1 = 0.0f, p2 = 0.0f;
    for (int d = tid; d < H2; d += 256) {
      p1 += label_att[c * H2 + d] * w1w[d];
      p2 += self_att[c * H2 + d] * w2w[d];
    }
    float dot1 = block_reduce_sum(p1, red);
    float dot2 = block_reduce_sum(p2, red);
    if (tid == 0) {
      float g1 = sigmoidf_(dot1 + w1b[0]);
      float g2 = sigmoidf_(dot2 + w2b[0]);
      g1n[c] = g1 / (g1 + g2);
    }
    __syncthreads();
  }

  for (int d = tid; d < H2; d += 256) {
    float s = 0.0f;
#pragma unroll
    for (int c = 0; c < NC; c++) {
      float g = g1n[c];
      s += g * label_att[c * H2 + d] + (1.0f - g) * self_att[c * H2 + d];
    }
    avg[d] = s * (1.0f / (float)NC);
  }
  __syncthreads();

  for (int j = 0; j < NC; j++) {
    float p = 0.0f;
    for (int d = tid; d < H2; d += 256) p += avg[d] * Wout[d * NC + j];
    float dot = block_reduce_sum(p, red);
    if (tid == 0) out[j] = sigmoidf_(dot + bout[j]);
    __syncthreads();
  }
}

// ---------------------------------------------------------------------------
extern "C" void kernel_launch(void* const* d_in, const int* in_sizes, int n_in,
                              void* d_out, int out_size, void* d_ws, size_t ws_size,
                              hipStream_t stream) {
  (void)in_sizes; (void)n_in; (void)out_size; (void)ws_size;
  const float* sent = (const float*)d_in[0];
  // d_in[1] sent_pool, d_in[2] abstract: unused by the reference
  const float* lab  = (const float*)d_in[3];
  const float* W1   = (const float*)d_in[4];
  const float* b1   = (const float*)d_in[5];
  const float* W2   = (const float*)d_in[6];
  const float* b2   = (const float*)d_in[7];
  const float* w1w  = (const float*)d_in[8];
  const float* w1b  = (const float*)d_in[9];
  const float* w2w  = (const float*)d_in[10];
  const float* w2b  = (const float*)d_in[11];
  const float* Wout = (const float*)d_in[12];
  const float* bout = (const float*)d_in[13];

  float* ws        = (float*)d_ws;
  float* scores    = ws;                        // SEQ*16 f32
  float* stats     = scores + (size_t)SEQ * NC; // 32 f32
  float* self_att  = stats + 32;                // 16*1024 f32
  float* label_att = self_att + NC * H2;        // 16*1024 f32
  __bf16* W1t  = (__bf16*)(label_att + NC * H2);   // 512*1024 bf16
  __bf16* W2t  = W1t + (size_t)DA * H2;            // 16*512 bf16
  __bf16* labb = W2t + NC * DA;                    // 16*512 bf16

  const int nprep = DA * H2 + 2 * NC * DA;
  k_prep<<<(nprep + 255) / 256, 256, 0, stream>>>(W1, W2, lab, W1t, W2t, labb);
  const int nzero = 2 * NC * H2;
  k_zero<<<(nzero + 255) / 256, 256, 0, stream>>>(self_att, nzero);
  k_gemm1<<<SEQ / 32, 256, 0, stream>>>(sent, W1t, b1, W2t, b2, scores);
  k_colstats<<<NC, 256, 0, stream>>>(scores, stats);
  k_selfatt<<<SEQ / 256, 256, 0, stream>>>(sent, scores, stats, self_att);
  k_labelatt<<<SEQ / 128, 256, 0, stream>>>(sent, labb, label_att);
  k_final<<<1, 256, 0, stream>>>(self_att, label_att, w1w, w1b, w2w, w2b, Wout, bout,
                                 (float*)d_out);
}
